// KANLayer_63660005261529
// MI455X (gfx1250) — compile-verified, matrix operand order checked by reading, not run on hardware
//
#include <hip/hip_runtime.h>

typedef __attribute__((ext_vector_type(16))) __bf16 v16bf;
typedef __attribute__((ext_vector_type(8)))  __bf16 v8bf;
typedef __attribute__((ext_vector_type(8)))  float  v8f;

#define BATCH 4096
#define IDIM  1024
#define ODIM  1024
#define KCP   5
#define KTOT  (IDIM * KCP)   // 5120 flattened reduction dim

// ---------------------------------------------------------------------------
// Kernel 1: cubic B-spline basis (Cox-de Boor, uniform knots on [-1,1], 9 knots)
// One thread per (b, i); writes 5 bf16 values -> basis[b][i*5 + k]
// ---------------------------------------------------------------------------
__global__ __launch_bounds__(256) void basis_kernel(const float* __restrict__ x,
                                                    __bf16* __restrict__ basis)
{
    const int idx = blockIdx.x * 256 + threadIdx.x;      // over BATCH*IDIM
    const int b = idx / IDIM;
    const int i = idx % IDIM;
    const float xv = x[idx];

    // degree-0 indicator basis over 8 spans; knots t_j = -1 + 0.25 j
    float Bv[8];
#pragma unroll
    for (int j = 0; j < 8; ++j) {
        const float t0 = -1.0f + 0.25f * (float)j;
        const float t1 = t0 + 0.25f;
        Bv[j] = (xv >= t0 && xv < t1) ? 1.0f : 0.0f;
    }
    // refinement to degree 3; all denominators are 0.25*r (uniform knots)
#pragma unroll
    for (int r = 1; r <= 3; ++r) {
        const float inv = 1.0f / (0.25f * (float)r);
#pragma unroll
        for (int j = 0; j < 7; ++j) {
            if (j < 8 - r) {
                const float tj   = -1.0f + 0.25f * (float)j;
                const float tjr1 = tj + 0.25f * (float)(r + 1);
                const float left  = (xv - tj)   * inv * Bv[j];
                const float right = (tjr1 - xv) * inv * Bv[j + 1];
                Bv[j] = left + right;
            }
        }
    }

    __bf16* dst = basis + (size_t)b * KTOT + (size_t)i * KCP;
#pragma unroll
    for (int k = 0; k < KCP; ++k) dst[k] = (__bf16)Bv[k];
}

// ---------------------------------------------------------------------------
// Kernel 2: fuse importance * coeffs and transpose to output-major bf16:
//   wT[o][i*5 + k] = importance[i,o] * coeffs[i,o,k]
// so the GEMM's B operand has the same per-lane layout as A.
// ---------------------------------------------------------------------------
__global__ __launch_bounds__(256) void weight_kernel(const float* __restrict__ coeffs,
                                                     const float* __restrict__ importance,
                                                     __bf16* __restrict__ wT)
{
    const int idx = blockIdx.x * 256 + threadIdx.x;      // over IDIM*ODIM
    const int i = idx / ODIM;
    const int o = idx % ODIM;
    const float imp = importance[idx];
    const float* c = coeffs + (size_t)idx * KCP;
    __bf16* dst = wT + (size_t)o * KTOT + (size_t)i * KCP;
#pragma unroll
    for (int k = 0; k < KCP; ++k) dst[k] = (__bf16)(imp * c[k]);
}

// ---------------------------------------------------------------------------
// Kernel 3: out[4096,1024] = A[4096,5120] x B[1024,5120]^T via
// v_wmma_f32_16x16x32_bf16.
//   block: 256 threads = 8 wave32, block tile 128(M) x 128(N)
//   waves: 2(M) x 4(N); each wave owns a 64x32 sub-tile = 4x2 WMMA fragments
//   -> per K-step(32): 12 b128 loads feed 8 WMMAs (1.5 loads/WMMA)
//
// Per-lane fragment addressing follows the CDNA5 16-bit A layout:
//   lanes 0-15  : row M=lane,    K = {kk+0..7,  kk+16..23}
//   lanes 16-31 : row M=lane-16, K = {kk+8..15, kk+24..31}
// B (stored N-major) uses the identical pattern with N in place of M.
// ---------------------------------------------------------------------------
__global__ __launch_bounds__(256) void kan_gemm(const __bf16* __restrict__ A,   // [BATCH][KTOT]
                                                const __bf16* __restrict__ Bm,  // [ODIM][KTOT]
                                                float* __restrict__ C)          // [BATCH][ODIM]
{
    const int lane  = threadIdx.x & 31;
    const int wave  = threadIdx.x >> 5;   // 0..7
    const int waveM = wave & 1;           // 0..1  -> 64 rows each
    const int waveN = wave >> 1;          // 0..3  -> 32 cols each
    const int half  = lane >> 4;          // 0/1
    const int ln    = lane & 15;

    const int mBase = blockIdx.y * 128 + waveM * 64;
    const int nBase = blockIdx.x * 128 + waveN * 32;
    const int kOff  = half * 8;

    const __bf16* aRow0 = A  + (size_t)(mBase + ln)      * KTOT + kOff;
    const __bf16* aRow1 = A  + (size_t)(mBase + 16 + ln) * KTOT + kOff;
    const __bf16* aRow2 = A  + (size_t)(mBase + 32 + ln) * KTOT + kOff;
    const __bf16* aRow3 = A  + (size_t)(mBase + 48 + ln) * KTOT + kOff;
    const __bf16* bRow0 = Bm + (size_t)(nBase + ln)      * KTOT + kOff;
    const __bf16* bRow1 = Bm + (size_t)(nBase + 16 + ln) * KTOT + kOff;

    v8f acc00 = {}, acc01 = {};
    v8f acc10 = {}, acc11 = {};
    v8f acc20 = {}, acc21 = {};
    v8f acc30 = {}, acc31 = {};

#pragma unroll 2
    for (int kk = 0; kk < KTOT; kk += 32) {
        union { v16bf v; v8bf h[2]; } a0, a1, a2, a3, b0, b1;
        b0.h[0] = *(const v8bf*)(bRow0 + kk);
        b0.h[1] = *(const v8bf*)(bRow0 + kk + 16);
        b1.h[0] = *(const v8bf*)(bRow1 + kk);
        b1.h[1] = *(const v8bf*)(bRow1 + kk + 16);
        a0.h[0] = *(const v8bf*)(aRow0 + kk);
        a0.h[1] = *(const v8bf*)(aRow0 + kk + 16);
        a1.h[0] = *(const v8bf*)(aRow1 + kk);
        a1.h[1] = *(const v8bf*)(aRow1 + kk + 16);
        a2.h[0] = *(const v8bf*)(aRow2 + kk);
        a2.h[1] = *(const v8bf*)(aRow2 + kk + 16);
        a3.h[0] = *(const v8bf*)(aRow3 + kk);
        a3.h[1] = *(const v8bf*)(aRow3 + kk + 16);

        // pull the next K-chunk toward the caches (global_prefetch_b8)
        __builtin_prefetch(aRow0 + kk + 64, 0, 1);
        __builtin_prefetch(bRow0 + kk + 64, 0, 1);

        acc00 = __builtin_amdgcn_wmma_f32_16x16x32_bf16(false, a0.v, false, b0.v,
                                                        (short)0, acc00, false, false);
        acc01 = __builtin_amdgcn_wmma_f32_16x16x32_bf16(false, a0.v, false, b1.v,
                                                        (short)0, acc01, false, false);
        acc10 = __builtin_amdgcn_wmma_f32_16x16x32_bf16(false, a1.v, false, b0.v,
                                                        (short)0, acc10, false, false);
        acc11 = __builtin_amdgcn_wmma_f32_16x16x32_bf16(false, a1.v, false, b1.v,
                                                        (short)0, acc11, false, false);
        acc20 = __builtin_amdgcn_wmma_f32_16x16x32_bf16(false, a2.v, false, b0.v,
                                                        (short)0, acc20, false, false);
        acc21 = __builtin_amdgcn_wmma_f32_16x16x32_bf16(false, a2.v, false, b1.v,
                                                        (short)0, acc21, false, false);
        acc30 = __builtin_amdgcn_wmma_f32_16x16x32_bf16(false, a3.v, false, b0.v,
                                                        (short)0, acc30, false, false);
        acc31 = __builtin_amdgcn_wmma_f32_16x16x32_bf16(false, a3.v, false, b1.v,
                                                        (short)0, acc31, false, false);
    }

    // Epilogue per CDNA5 32-bit C/D layout: VGPR r holds M = r + 8*half, N = ln.
#pragma unroll
    for (int r = 0; r < 8; ++r) {
        const int m0 = mBase + r + half * 8;
        C[(size_t)m0        * ODIM + (nBase + ln)]      = acc00[r];
        C[(size_t)m0        * ODIM + (nBase + 16 + ln)] = acc01[r];
        C[(size_t)(m0 + 16) * ODIM + (nBase + ln)]      = acc10[r];
        C[(size_t)(m0 + 16) * ODIM + (nBase + 16 + ln)] = acc11[r];
        C[(size_t)(m0 + 32) * ODIM + (nBase + ln)]      = acc20[r];
        C[(size_t)(m0 + 32) * ODIM + (nBase + 16 + ln)] = acc21[r];
        C[(size_t)(m0 + 48) * ODIM + (nBase + ln)]      = acc30[r];
        C[(size_t)(m0 + 48) * ODIM + (nBase + 16 + ln)] = acc31[r];
    }
}

// ---------------------------------------------------------------------------
extern "C" void kernel_launch(void* const* d_in, const int* in_sizes, int n_in,
                              void* d_out, int out_size, void* d_ws, size_t ws_size,
                              hipStream_t stream)
{
    const float* x          = (const float*)d_in[0];   // [4096,1024]
    const float* coeffs     = (const float*)d_in[1];   // [1024,1024,5]
    const float* importance = (const float*)d_in[2];   // [1024,1024]
    float* out = (float*)d_out;                        // [4096,1024]

    __bf16* basis = (__bf16*)d_ws;                                     // 40 MB
    __bf16* wT    = (__bf16*)((char*)d_ws +
                              (size_t)BATCH * KTOT * sizeof(__bf16));  // +10 MB

    basis_kernel <<<(BATCH * IDIM) / 256, 256, 0, stream>>>(x, basis);
    weight_kernel<<<(IDIM * ODIM) / 256, 256, 0, stream>>>(coeffs, importance, wT);

    dim3 grid(ODIM / 128, BATCH / 128);                // (8, 32) blocks
    kan_gemm<<<grid, 256, 0, stream>>>(basis, wT, out);
}
